// MambaSSMBlock_69741678953030
// MI455X (gfx1250) — matrix operand order repaired and checked by reading, not compiled, not run on hardware
//
#include <hip/hip_runtime.h>
#include <hip/hip_bf16.h>
#include <math.h>

// ---------------- problem constants ----------------
#define D_MODEL 896
#define D_STATE 16
#define D_INNER 1792
#define DT_RANK 56
#define BB 2
#define TT 2048
#define NROWS (BB * TT)            // 4096 token rows
#define XZ_N (2 * D_INNER)         // 3584
#define XDBL_N (DT_RANK + 2 * D_STATE) // 88
#define DTR_PAD 64                 // dt_rank padded to WMMA K multiple

typedef __bf16 bf16;
typedef __attribute__((ext_vector_type(16))) __bf16 v16bf;
typedef __attribute__((ext_vector_type(8)))  float  v8f;

union FragU { v16bf bf; uint4 u4[2]; };

// A-matrix fragment (16x32, 16-bit): lane L holds row (L&15); K chunks
// [k + 8*half, +8) and [k + 16 + 8*half, +8)  (half = L>>4).
__device__ __forceinline__ v16bf ld_fragA(const bf16* base, int ld, int row, int k, int half) {
    const bf16* q = base + (size_t)row * ld + k + half * 8;
    FragU f;
    f.u4[0] = *(const uint4*)(q);
    f.u4[1] = *(const uint4*)(q + 16);
    return f.bf;
}

// B-matrix fragment (32x16, 16-bit): lane L holds column (L&15); K chunks
// [k + 8*half, +8) and [k + 16 + 8*half, +8)  (half = L>>4), matching A.
// Source W is row-major [N x K].
__device__ __forceinline__ v16bf ld_fragB(const bf16* base, int ld, int col, int k, int half) {
    const bf16* q = base + (size_t)col * ld + k + half * 8;
    FragU f;
    f.u4[0] = *(const uint4*)(q);
    f.u4[1] = *(const uint4*)(q + 16);
    return f.bf;
}

__device__ __forceinline__ v16bf zero_frag() {
    FragU f;
    f.u4[0] = make_uint4(0u, 0u, 0u, 0u);
    f.u4[1] = make_uint4(0u, 0u, 0u, 0u);
    return f.bf;
}

__device__ __forceinline__ v8f wmma_bf16(v16bf a, v16bf b, v8f c) {
    // (neg_a, A, neg_b, B, c_mod, C, reuse_a, reuse_b)
    return __builtin_amdgcn_wmma_f32_16x16x32_bf16(false, a, false, b, (short)0, c, false, false);
}

__device__ __forceinline__ float fsilu(float x) { return x / (1.f + __expf(-x)); }

// ---------------- kernel 0: weight prep / casts ----------------
__global__ void mamba_prep_kernel(const float* __restrict__ w_in, const float* __restrict__ w_xp,
                                  const float* __restrict__ w_dt, const float* __restrict__ w_out,
                                  const float* __restrict__ A_log,
                                  bf16* __restrict__ w_in_b, bf16* __restrict__ w_xp_b,
                                  bf16* __restrict__ w_dt_b, bf16* __restrict__ w_out_b,
                                  float* __restrict__ A_neg, bf16* __restrict__ dtr) {
    int i = blockIdx.x * 256 + threadIdx.x;
    const int n1 = XZ_N * D_MODEL;        // in_proj_w
    const int n2 = XDBL_N * D_INNER;      // x_proj_w
    const int n3 = D_INNER * DTR_PAD;     // dt_proj_w (padded K)
    const int n4 = D_MODEL * D_INNER;     // out_proj_w
    const int n5 = D_INNER * D_STATE;     // A
    const int n6 = NROWS * DTR_PAD;       // zero dt_r pad buffer
    if (i < n1) { w_in_b[i] = (bf16)w_in[i]; return; } i -= n1;
    if (i < n2) { w_xp_b[i] = (bf16)w_xp[i]; return; } i -= n2;
    if (i < n3) {
        int r = i / DTR_PAD, c = i % DTR_PAD;
        w_dt_b[i] = (c < DT_RANK) ? (bf16)w_dt[r * DT_RANK + c] : (bf16)0.f;
        return;
    } i -= n3;
    if (i < n4) { w_out_b[i] = (bf16)w_out[i]; return; } i -= n4;
    if (i < n5) { A_neg[i] = -__expf(A_log[i]); return; } i -= n5;
    if (i < n6) { dtr[i] = (bf16)0.f; }
}

// ---------------- kernel 1: RMSNorm -> bf16 ----------------
__global__ void mamba_rmsnorm_kernel(const float* __restrict__ x, const float* __restrict__ w,
                                     bf16* __restrict__ xn) {
    int row = blockIdx.x;
    __shared__ float red[8];
    float s = 0.f;
    for (int c = threadIdx.x; c < D_MODEL; c += 256) {
        float v = x[(size_t)row * D_MODEL + c];
        s += v * v;
    }
    for (int m = 16; m >= 1; m >>= 1) s += __shfl_xor(s, m, 32);
    if ((threadIdx.x & 31) == 0) red[threadIdx.x >> 5] = s;
    __syncthreads();
    float tot = 0.f;
    #pragma unroll
    for (int i = 0; i < 8; ++i) tot += red[i];
    float scale = rsqrtf(tot / (float)D_MODEL + 1e-6f);
    for (int c = threadIdx.x; c < D_MODEL; c += 256) {
        xn[(size_t)row * D_MODEL + c] = (bf16)(x[(size_t)row * D_MODEL + c] * scale * w[c]);
    }
}

// ---------------- 32x64 wave-tile WMMA GEMM core ----------------
// Computes C[32x64] tile at (m0, n0): C = A[M x K] * W[N x K]^T, bf16 in, f32 acc.
// acc[h][q]: h = M half (0: m0.., 1: m0+16..), q = N quarter (n0 + 16q).
struct Tile64Acc { v8f c[2][4]; };

__device__ __forceinline__ void gemm_tile_32x64(const bf16* __restrict__ A, int lda,
                                                const bf16* __restrict__ W, int ldw,
                                                int Kdim, int m0, int n0, int r, int half,
                                                Tile64Acc& acc) {
    #pragma unroll
    for (int h = 0; h < 2; ++h)
        #pragma unroll
        for (int q = 0; q < 4; ++q)
            acc.c[h][q] = (v8f){};
    for (int k = 0; k < Kdim; k += 32) {
        v16bf a0 = ld_fragA(A, lda, m0 + r, k, half);
        v16bf a1 = ld_fragA(A, lda, m0 + 16 + r, k, half);
        v16bf b0 = ld_fragB(W, ldw, n0 + r, k, half);
        v16bf b1 = ld_fragB(W, ldw, n0 + 16 + r, k, half);
        v16bf b2 = ld_fragB(W, ldw, n0 + 32 + r, k, half);
        v16bf b3 = ld_fragB(W, ldw, n0 + 48 + r, k, half);
        acc.c[0][0] = wmma_bf16(a0, b0, acc.c[0][0]);
        acc.c[0][1] = wmma_bf16(a0, b1, acc.c[0][1]);
        acc.c[0][2] = wmma_bf16(a0, b2, acc.c[0][2]);
        acc.c[0][3] = wmma_bf16(a0, b3, acc.c[0][3]);
        acc.c[1][0] = wmma_bf16(a1, b0, acc.c[1][0]);
        acc.c[1][1] = wmma_bf16(a1, b1, acc.c[1][1]);
        acc.c[1][2] = wmma_bf16(a1, b2, acc.c[1][2]);
        acc.c[1][3] = wmma_bf16(a1, b3, acc.c[1][3]);
    }
}

// ---------------- kernel 2: in_proj GEMM (M=4096,N=3584,K=896) -> bf16 ----------------
__global__ void mamba_gemm_inproj(const bf16* __restrict__ A, const bf16* __restrict__ Bw,
                                  bf16* __restrict__ Cout) {
    const int TN = XZ_N / 64;  // 56
    int tile = blockIdx.x * 8 + (threadIdx.x >> 5);
    if (tile >= (NROWS / 32) * TN) return;
    int m0 = (tile / TN) * 32, n0 = (tile % TN) * 64;
    int lane = threadIdx.x & 31, r = lane & 15, half = lane >> 4;
    Tile64Acc acc;
    gemm_tile_32x64(A, D_MODEL, Bw, D_MODEL, D_MODEL, m0, n0, r, half, acc);
    #pragma unroll
    for (int h = 0; h < 2; ++h)
        #pragma unroll
        for (int q = 0; q < 4; ++q)
            #pragma unroll
            for (int v = 0; v < 8; ++v) {
                int row = m0 + 16 * h + half * 8 + v;
                Cout[(size_t)row * XZ_N + n0 + 16 * q + r] = (bf16)acc.c[h][q][v];
            }
}

// ---------------- kernel 3: depthwise causal conv(K=4) + SiLU ----------------
__global__ void mamba_conv_silu(const bf16* __restrict__ xz, const float* __restrict__ cw,
                                const float* __restrict__ cb, bf16* __restrict__ u) {
    int idx = blockIdx.x * 256 + threadIdx.x;
    if (idx >= NROWS * D_INNER) return;
    int d = idx % D_INNER;
    int row = idx / D_INNER;
    int t = row % TT;
    float acc = cb[d];
    #pragma unroll
    for (int j = 0; j < 4; ++j) {
        int tt = t - 3 + j;
        if (tt >= 0)
            acc += cw[d * 4 + j] * (float)xz[(size_t)(row - 3 + j) * XZ_N + d];
    }
    u[idx] = (bf16)fsilu(acc);
}

// ---------------- kernel 4: x_proj GEMM (N=88, col-guarded, 32x16 tiles) ----------------
__global__ void mamba_gemm_xproj(const bf16* __restrict__ A, const bf16* __restrict__ Bw,
                                 bf16* __restrict__ dtr, float* __restrict__ Bm,
                                 float* __restrict__ Cm) {
    const int TN = 6;  // ceil(88/16)
    int tile = blockIdx.x * 8 + (threadIdx.x >> 5);
    if (tile >= (NROWS / 32) * TN) return;
    int m0 = (tile / TN) * 32, n0 = (tile % TN) * 16;
    int lane = threadIdx.x & 31, r = lane & 15, half = lane >> 4;
    int ecol = n0 + r;
    v8f c0 = {}, c1 = {};
    for (int k = 0; k < D_INNER; k += 32) {
        v16bf a0 = ld_fragA(A, D_INNER, m0 + r, k, half);
        v16bf a1 = ld_fragA(A, D_INNER, m0 + 16 + r, k, half);
        v16bf b0 = (ecol < XDBL_N) ? ld_fragB(Bw, D_INNER, ecol, k, half) : zero_frag();
        c0 = wmma_bf16(a0, b0, c0);
        c1 = wmma_bf16(a1, b0, c1);
    }
    if (ecol >= XDBL_N) return;
    #pragma unroll
    for (int v = 0; v < 8; ++v) {
        int row0 = m0 + half * 8 + v;
        float v0 = c0[v], v1 = c1[v];
        if (ecol < DT_RANK) {
            dtr[(size_t)row0 * DTR_PAD + ecol] = (bf16)v0;
            dtr[(size_t)(row0 + 16) * DTR_PAD + ecol] = (bf16)v1;
        } else if (ecol < DT_RANK + D_STATE) {
            Bm[(size_t)row0 * D_STATE + (ecol - DT_RANK)] = v0;
            Bm[(size_t)(row0 + 16) * D_STATE + (ecol - DT_RANK)] = v1;
        } else {
            Cm[(size_t)row0 * D_STATE + (ecol - DT_RANK - D_STATE)] = v0;
            Cm[(size_t)(row0 + 16) * D_STATE + (ecol - DT_RANK - D_STATE)] = v1;
        }
    }
}

// ---------------- kernel 5: dt_proj GEMM (K=64 padded) + softplus ----------------
__global__ void mamba_gemm_dtproj(const bf16* __restrict__ A, const bf16* __restrict__ Bw,
                                  const float* __restrict__ bias, float* __restrict__ dt) {
    const int TN = D_INNER / 64;  // 28
    int tile = blockIdx.x * 8 + (threadIdx.x >> 5);
    if (tile >= (NROWS / 32) * TN) return;
    int m0 = (tile / TN) * 32, n0 = (tile % TN) * 64;
    int lane = threadIdx.x & 31, r = lane & 15, half = lane >> 4;
    Tile64Acc acc;
    gemm_tile_32x64(A, DTR_PAD, Bw, DTR_PAD, DTR_PAD, m0, n0, r, half, acc);
    #pragma unroll
    for (int q = 0; q < 4; ++q) {
        float bq = bias[n0 + 16 * q + r];
        #pragma unroll
        for (int h = 0; h < 2; ++h)
            #pragma unroll
            for (int v = 0; v < 8; ++v) {
                int row = m0 + 16 * h + half * 8 + v;
                float xv = acc.c[h][q][v] + bq;
                xv = (xv > 20.f) ? xv : log1pf(__expf(xv));
                dt[(size_t)row * D_INNER + n0 + 16 * q + r] = xv;
            }
    }
}

// ---------------- kernel 6: selective scan (lane per (b,d,s)) ----------------
__global__ void mamba_scan_kernel(const float* __restrict__ dt, const bf16* __restrict__ u,
                                  const float* __restrict__ Bm, const float* __restrict__ Cm,
                                  const float* __restrict__ A_neg, const float* __restrict__ Dp,
                                  const bf16* __restrict__ xz, bf16* __restrict__ y) {
    int gid = blockIdx.x * 256 + threadIdx.x;   // exactly B*D_INNER*16 threads
    int s = gid & 15;
    int ch = gid >> 4;
    int d = ch % D_INNER;
    int b = ch / D_INNER;
    float a = A_neg[d * D_STATE + s];
    float Dv = Dp[d];
    float h = 0.f;
    size_t row = (size_t)b * TT;
    for (int t = 0; t < TT; ++t, ++row) {
        float dtv = dt[row * D_INNER + d];
        float uv = (float)u[row * D_INNER + d];
        float Bv = Bm[row * D_STATE + s];
        float Cv = Cm[row * D_STATE + s];
        h = __expf(dtv * a) * h + dtv * Bv * uv;
        float p = h * Cv;
        p += __shfl_xor(p, 1, 32);
        p += __shfl_xor(p, 2, 32);
        p += __shfl_xor(p, 4, 32);
        p += __shfl_xor(p, 8, 32);
        if (s == 0) {
            float z = (float)xz[row * XZ_N + D_INNER + d];
            float yy = (p + uv * Dv) * fsilu(z);
            y[row * D_INNER + d] = (bf16)yy;
        }
    }
}

// ---------------- kernel 7: out_proj GEMM + residual ----------------
__global__ void mamba_gemm_outproj(const bf16* __restrict__ A, const bf16* __restrict__ Bw,
                                   const float* __restrict__ resid, float* __restrict__ out) {
    const int TN = D_MODEL / 64;  // 14
    int tile = blockIdx.x * 8 + (threadIdx.x >> 5);
    if (tile >= (NROWS / 32) * TN) return;
    int m0 = (tile / TN) * 32, n0 = (tile % TN) * 64;
    int lane = threadIdx.x & 31, r = lane & 15, half = lane >> 4;
    Tile64Acc acc;
    gemm_tile_32x64(A, D_INNER, Bw, D_INNER, D_INNER, m0, n0, r, half, acc);
    #pragma unroll
    for (int h = 0; h < 2; ++h)
        #pragma unroll
        for (int q = 0; q < 4; ++q)
            #pragma unroll
            for (int v = 0; v < 8; ++v) {
                int row = m0 + 16 * h + half * 8 + v;
                size_t i = (size_t)row * D_MODEL + n0 + 16 * q + r;
                out[i] = acc.c[h][q][v] + resid[i];
            }
}

// ---------------- host launch ----------------
extern "C" void kernel_launch(void* const* d_in, const int* in_sizes, int n_in,
                              void* d_out, int out_size, void* d_ws, size_t ws_size,
                              hipStream_t stream) {
    const float* x        = (const float*)d_in[0];
    const float* norm_w   = (const float*)d_in[1];
    const float* in_proj  = (const float*)d_in[2];
    const float* conv_w   = (const float*)d_in[3];
    const float* conv_b   = (const float*)d_in[4];
    const float* x_proj   = (const float*)d_in[5];
    const float* dt_proj  = (const float*)d_in[6];
    const float* dt_bias  = (const float*)d_in[7];
    const float* A_log    = (const float*)d_in[8];
    const float* Dp       = (const float*)d_in[9];
    const float* out_proj = (const float*)d_in[10];
    float* out = (float*)d_out;

    char* p = (char*)d_ws;
    auto alloc = [&](size_t bytes) -> void* {
        void* r = (void*)p;
        p += (bytes + 255) & ~(size_t)255;
        return r;
    };
    bf16* w_in_b  = (bf16*)alloc((size_t)XZ_N * D_MODEL * 2);
    bf16* w_xp_b  = (bf16*)alloc((size_t)XDBL_N * D_INNER * 2);
    bf16* w_dt_b  = (bf16*)alloc((size_t)D_INNER * DTR_PAD * 2);
    bf16* w_out_b = (bf16*)alloc((size_t)D_MODEL * D_INNER * 2);
    float* A_neg  = (float*)alloc((size_t)D_INNER * D_STATE * 4);
    bf16* xn      = (bf16*)alloc((size_t)NROWS * D_MODEL * 2);
    bf16* xz      = (bf16*)alloc((size_t)NROWS * XZ_N * 2);
    bf16* u       = (bf16*)alloc((size_t)NROWS * D_INNER * 2);
    bf16* dtr     = (bf16*)alloc((size_t)NROWS * DTR_PAD * 2);
    float* Bm     = (float*)alloc((size_t)NROWS * D_STATE * 4);
    float* Cm     = (float*)alloc((size_t)NROWS * D_STATE * 4);
    float* dt     = (float*)alloc((size_t)NROWS * D_INNER * 4);
    bf16* yb      = (bf16*)alloc((size_t)NROWS * D_INNER * 2);
    (void)ws_size; (void)in_sizes; (void)n_in; (void)out_size;

    // 0) weight casts / padding / A = -exp(A_log)
    {
        int total = XZ_N * D_MODEL + XDBL_N * D_INNER + D_INNER * DTR_PAD +
                    D_MODEL * D_INNER + D_INNER * D_STATE + NROWS * DTR_PAD;
        int blocks = (total + 255) / 256;
        mamba_prep_kernel<<<blocks, 256, 0, stream>>>(in_proj, x_proj, dt_proj, out_proj, A_log,
                                                      w_in_b, w_xp_b, w_dt_b, w_out_b, A_neg, dtr);
    }
    // 1) RMSNorm
    mamba_rmsnorm_kernel<<<NROWS, 256, 0, stream>>>(x, norm_w, xn);
    // 2) in_proj GEMM -> xz
    mamba_gemm_inproj<<<(NROWS / 32) * (XZ_N / 64) / 8, 256, 0, stream>>>(xn, w_in_b, xz);
    // 3) conv + SiLU -> u
    mamba_conv_silu<<<(NROWS * D_INNER) / 256, 256, 0, stream>>>(xz, conv_w, conv_b, u);
    // 4) x_proj GEMM -> dt_r / B / C
    mamba_gemm_xproj<<<(NROWS / 32) * 6 / 8, 256, 0, stream>>>(u, w_xp_b, dtr, Bm, Cm);
    // 5) dt_proj GEMM + softplus -> dt
    mamba_gemm_dtproj<<<(NROWS / 32) * (D_INNER / 64) / 8, 256, 0, stream>>>(dtr, w_dt_b, dt_bias, dt);
    // 6) selective scan + gating -> y (bf16)
    mamba_scan_kernel<<<(BB * D_INNER * D_STATE) / 256, 256, 0, stream>>>(dt, u, Bm, Cm, A_neg, Dp, xz, yb);
    // 7) out_proj GEMM + residual -> out
    mamba_gemm_outproj<<<(NROWS / 32) * (D_MODEL / 64) / 8, 256, 0, stream>>>(yb, w_out_b, x, out);
}
